// GCNLayer_35476429865413
// MI455X (gfx1250) — matrix operand order repaired and checked by reading, not compiled, not run on hardware
//
#include <hip/hip_runtime.h>

// ---------------------------------------------------------------------------
// GCN forward on MI455X (gfx1250, wave32):
//   h      = relu(adj @ (feats @ W1) + b1)
//   logits = adj @ (h @ W2) + b2
//   out    = log_softmax(logits, axis=1)
//
// v_wmma_f32_16x16x32_bf16 core, f32 accumulate. adj (256 MB fp32) streams
// from HBM exactly once per adj-GEMM. B operand staged into LDS by the Tensor
// Data Mover, TRIPLE-buffered with two TDMs in flight (TENSORcnt-synced).
// Raw s_barrier_signal/wait (no vmem drain) keeps the A-stream prefetch alive
// across slab boundaries; manual two-substep ping-pong removes register
// rotation copies and WMMA->VALU hazard NOPs.
// ---------------------------------------------------------------------------

typedef __bf16 bf16_t;
typedef __attribute__((ext_vector_type(16))) __bf16 v16bf;
typedef __attribute__((ext_vector_type(8)))  __bf16 v8bf;
typedef __attribute__((ext_vector_type(8)))  float  v8f;
typedef __attribute__((ext_vector_type(4)))  unsigned int u32x4;
typedef __attribute__((ext_vector_type(8)))  int i32x8;
typedef __attribute__((ext_vector_type(4)))  int i32x4;

#define GCN_N    8192
#define GCN_DIN  512
#define GCN_DHID 256
#define GCN_DOUT 64

static __device__ __forceinline__ unsigned short bf16_bits(float f) {
    __bf16 h = (__bf16)f;
    unsigned short b;
    __builtin_memcpy(&b, &h, 2);
    return b;
}

// Workgroup barrier without the __syncthreads() vmem drain: TDM->LDS
// visibility is ordered by s_wait_tensorcnt on the issuing wave; we only need
// the control barrier + a compile-time reordering fence for LDS accesses.
static __device__ __forceinline__ void block_sync_light() {
    asm volatile("" ::: "memory");
    __builtin_amdgcn_s_barrier_signal(-1);
    __builtin_amdgcn_s_barrier_wait(-1);
    asm volatile("" ::: "memory");
}

// ---------------------------------------------------------------------------
// TDM: load a 2D tile of BT (rows = B columns, contiguous K) into LDS.
//   tile = COLS rows x KSTEP bf16 elements, rows packed at KSTEP*2 bytes.
// D# per CDNA5 ISA; all fields block-uniform -> SGPRs.
// ---------------------------------------------------------------------------
template <int COLS, int KSTEP>
static __device__ __forceinline__ void tdm_issue_b(unsigned long long gaddr,
                                                   unsigned ldsOff, int K) {
    const unsigned TD0 = 1u << 30;   // tensor_dim0: huge (tiles always in-bounds)
    const unsigned TD1 = 1u << 20;   // tensor_dim1: huge

    u32x4 g0;
    g0[0] = 1u;                                        // count=1, user descriptor
    g0[1] = ldsOff;                                    // lds_addr (bytes)
    g0[2] = (unsigned)gaddr;                           // global_addr[31:0]
    g0[3] = ((unsigned)(gaddr >> 32) & 0x01FFFFFFu)    // global_addr[56:32]
            | 0x80000000u;                             // type=2 ("image")

    i32x8 g1;
    g1[0] = 0x00010000;                                // data_size=1 (2 bytes)
    g1[1] = (int)((TD0 & 0xFFFFu) << 16);              // tensor_dim0[15:0]
    g1[2] = (int)(((TD0 >> 16) & 0xFFFFu) |
                  ((TD1 & 0xFFFFu) << 16));            // dim0 hi | dim1 lo
    g1[3] = (int)(((TD1 >> 16) & 0xFFFFu) |
                  ((unsigned)KSTEP << 16));            // dim1 hi | tile_dim0
    g1[4] = COLS;                                      // tile_dim1 (tile_dim2=0)
    g1[5] = K;                                         // tensor_dim0_stride lo
    g1[6] = 0;                                         // stride hi | dim1_stride lo
    g1[7] = 0;

    i32x4 z4 = {};
#if defined(__clang_major__) && __clang_major__ >= 23
    i32x8 z8 = {};
    __builtin_amdgcn_tensor_load_to_lds(g0, g1, z4, z4, z8, 0);
#else
    __builtin_amdgcn_tensor_load_to_lds(g0, g1, z4, z4, 0);
#endif
}

// ---------------------------------------------------------------------------
// Small transpose+convert: out[n*K + k] = bf16(in[k*N + n])   (in is K x N f32)
// ---------------------------------------------------------------------------
__global__ void transpose_to_bf16(const float* __restrict__ in,
                                  bf16_t* __restrict__ out, int K, int N) {
    int idx = blockIdx.x * blockDim.x + threadIdx.x;
    if (idx >= K * N) return;
    int k = idx / N, n = idx - k * N;
    out[(size_t)n * K + k] = (bf16_t)in[idx];
}

// ---------------------------------------------------------------------------
// WMMA GEMM: C[M x Ncols] = A[M x K] @ B[K x Ncols]
//   Block: 256 threads = 8 waves = (8/CW) row-waves x CW col-waves.
//   Wave tile: 16 rows x NT*16 cols; block tile: (8/CW)*16 x CW*NT*16.
//   A: row-major f32 (cvt in-register) or bf16, streamed + ping-pong register
//      prefetch. B: TRANSPOSED bf16 (BT[Ncols x K]); KSTEP K-slabs staged to
//      LDS by the TDM, 3 buffers / 2 TDMs in flight; ds reads pipelined.
//   EPI 0: store bf16 TRANSPOSED (ldo = M)      [feeds next GEMM's B]
//   EPI 1: +bias, relu, store bf16 row-major    [H]
//   EPI 2: +bias, store f32 row-major           [logits]
// ---------------------------------------------------------------------------
template <int NT, int CW, int KSTEP, bool A_F32, int EPI>
__launch_bounds__(256)
__global__ void wmma_gemm(const void* __restrict__ Aptr,
                          const bf16_t* __restrict__ BT,
                          const float* __restrict__ bias,
                          void* __restrict__ Out,
                          int K, int ldo) {
    extern __shared__ char smem[];
    constexpr int COLS  = CW * NT * 16;          // block tile columns
    constexpr int BROWS = (8 / CW) * 16;         // block tile rows
    constexpr int BUFB  = COLS * KSTEP * 2;      // bytes per LDS buffer
    constexpr int SUB   = KSTEP / 32;            // 32-K sub-steps per slab
    constexpr int NBUF  = 3;                     // LDS buffers (2 TDMs in flight)

    const int lane  = threadIdx.x & 31;
    const int wave  = threadIdx.x >> 5;
    const int l15   = lane & 15;
    const int khalf = lane >> 4;                 // 0: low K-half, 1: high K-half

    const int rowW  = wave / CW;
    const int colW  = wave % CW;
    const int mW    = blockIdx.y * BROWS + rowW * 16;     // wave's first row
    const int nBase = blockIdx.x * COLS;                  // block's first col
    const int nCol0 = colW * NT * 16;                     // wave's col in LDS tile
    const int aRow  = mW + l15;                           // this lane's A row

    v8f acc[NT] = {};

    const float*  Af  = (const float*)Aptr;
    const bf16_t* Abf = (const bf16_t*)Aptr;

    const unsigned long long gbase =
        (unsigned long long)(uintptr_t)(BT + (size_t)nBase * K);

    const int nsub   = K / 32;                   // always even here
    const int nsteps = K / KSTEP;

    // Raw A chunks: two independent register sets (ping-pong, no copies).
    v8f  f00, f01, f10, f11;
    v8bf h00, h01, h10, h11;

    auto loadA = [&](int j, v8f& f0, v8f& f1, v8bf& h0, v8bf& h1) {
        const int k = j * 32 + khalf * 16;
        if constexpr (A_F32) {
            const float* pa = Af + (size_t)aRow * K + k;
            f0 = *(const v8f*)pa;
            f1 = *(const v8f*)(pa + 8);
        } else {
            const bf16_t* pa = Abf + (size_t)aRow * K + k;
            h0 = *(const v8bf*)pa;
            h1 = *(const v8bf*)(pa + 8);
        }
    };

    // Slab boundary: retire oldest TDM, barrier, launch TDM for slab i+2.
    auto slab_logic = [&](int j) {
        if ((j & (SUB - 1)) == 0) {
            const int i = j / SUB;
            if (wave == 0) {
                if (i + 1 < nsteps) __builtin_amdgcn_s_wait_tensorcnt(1);
                else                __builtin_amdgcn_s_wait_tensorcnt(0);
            }
            block_sync_light();
            if (wave == 0 && i + 2 < nsteps)
                tdm_issue_b<COLS, KSTEP>(
                    gbase + (unsigned long long)(i + 2) * KSTEP * 2,
                    (unsigned)(((i + 2) % NBUF) * BUFB), K);
        }
    };

    auto compute = [&](int j, v8f& f0, v8f& f1, v8bf& h0, v8bf& h1) {
        // Assemble A fragment (contiguous k = 16*khalf + s, same order as B).
        v16bf a;
        if constexpr (A_F32) {
#pragma unroll
            for (int q = 0; q < 8; ++q) { a[q] = (__bf16)f0[q]; a[q + 8] = (__bf16)f1[q]; }
        } else {
#pragma unroll
            for (int q = 0; q < 8; ++q) { a[q] = h0[q]; a[q + 8] = h1[q]; }
        }
        const char* bbuf = smem + ((j / SUB) % NBUF) * BUFB;
        const int   s    = j & (SUB - 1);
        auto ldsB = [&](int t) -> v16bf {
            return *(const v16bf*)(bbuf + (nCol0 + t * 16 + l15) * (KSTEP * 2) +
                                   s * 64 + khalf * 32);
        };
        // B-fragment pipelined WMMA chain: load t+1 while t computes.
        v16bf bcur = ldsB(0);
#pragma unroll
        for (int t = 0; t < NT; ++t) {
            v16bf bnext = bcur;
            if (t + 1 < NT) bnext = ldsB(t + 1);
            acc[t] = __builtin_amdgcn_wmma_f32_16x16x32_bf16(
                false, a, false, bcur, (short)0, acc[t], false, false);
            bcur = bnext;
        }
    };

    // Prologue: two TDMs in flight, first A chunk loading.
    if (wave == 0) {
        tdm_issue_b<COLS, KSTEP>(gbase, 0u, K);
        if (nsteps > 1)
            tdm_issue_b<COLS, KSTEP>(gbase + (unsigned long long)KSTEP * 2,
                                     (unsigned)BUFB, K);
    }
    loadA(0, f00, f01, h00, h01);

    for (int j = 0; j < nsub; j += 2) {
        slab_logic(j);
        loadA(j + 1, f10, f11, h10, h11);
        compute(j, f00, f01, h00, h01);

        slab_logic(j + 1);
        if (j + 2 < nsub) loadA(j + 2, f00, f01, h00, h01);
        compute(j + 1, f10, f11, h10, h11);
    }

    // ---- Epilogue. C/D layout: lane holds (M = r + 8*khalf, N = l15), r=0..7.
#pragma unroll
    for (int t = 0; t < NT; ++t) {
        const int n = nBase + nCol0 + t * 16 + l15;
        if (EPI == 0) {
            bf16_t* o = (bf16_t*)Out + (size_t)n * ldo + mW + khalf * 8;
#pragma unroll
            for (int r = 0; r < 8; r += 2) {
                unsigned pk = (unsigned)bf16_bits(acc[t][r]) |
                              ((unsigned)bf16_bits(acc[t][r + 1]) << 16);
                *(unsigned*)(o + r) = pk;
            }
        } else if (EPI == 1) {
            const float bv = bias[n];
#pragma unroll
            for (int r = 0; r < 8; ++r) {
                int   m = mW + khalf * 8 + r;
                float v = acc[t][r] + bv;
                v = v > 0.0f ? v : 0.0f;
                ((bf16_t*)Out)[(size_t)m * ldo + n] = (bf16_t)v;
            }
        } else {
            const float bv = bias[n];
#pragma unroll
            for (int r = 0; r < 8; ++r) {
                int m = mW + khalf * 8 + r;
                ((float*)Out)[(size_t)m * ldo + n] = acc[t][r] + bv;
            }
        }
    }
}

// ---------------------------------------------------------------------------
// Row-wise log-softmax over 64 columns; one wave32 per row, 2 cols per lane.
// ---------------------------------------------------------------------------
__global__ void log_softmax_rows(const float* __restrict__ logits,
                                 float* __restrict__ out) {
    const int row = blockIdx.x;
    const int l   = threadIdx.x;  // 0..31
    const float* x = logits + (size_t)row * GCN_DOUT;

    float x0 = x[l], x1 = x[l + 32];
    float m = fmaxf(x0, x1);
#pragma unroll
    for (int o = 16; o > 0; o >>= 1) m = fmaxf(m, __shfl_xor(m, o, 32));
    float s = __expf(x0 - m) + __expf(x1 - m);
#pragma unroll
    for (int o = 16; o > 0; o >>= 1) s += __shfl_xor(s, o, 32);
    const float lse = m + __logf(s);

    float* y = out + (size_t)row * GCN_DOUT;
    y[l]      = x0 - lse;
    y[l + 32] = x1 - lse;
}

// ---------------------------------------------------------------------------
extern "C" void kernel_launch(void* const* d_in, const int* in_sizes, int n_in,
                              void* d_out, int out_size, void* d_ws, size_t ws_size,
                              hipStream_t stream) {
    const float* feats = (const float*)d_in[0];  // [8192, 512]
    const float* adj   = (const float*)d_in[1];  // [8192, 8192]
    const float* W1    = (const float*)d_in[2];  // [512, 256]
    const float* b1    = (const float*)d_in[3];  // [256]
    const float* W2    = (const float*)d_in[4];  // [256, 64]
    const float* b2    = (const float*)d_in[5];  // [64]

    char*  ws  = (char*)d_ws;
    size_t off = 0;
    auto alloc = [&](size_t bytes) {
        void* p = ws + off;
        off = (off + bytes + 255) & ~(size_t)255;
        return p;
    };

    bf16_t* W1T    = (bf16_t*)alloc((size_t)GCN_DHID * GCN_DIN * 2);   // [256][512]
    bf16_t* W2T    = (bf16_t*)alloc((size_t)GCN_DOUT * GCN_DHID * 2);  // [64][256]
    bf16_t* X1T    = (bf16_t*)alloc((size_t)GCN_DHID * GCN_N * 2);     // [256][8192]
    bf16_t* H      = (bf16_t*)alloc((size_t)GCN_N * GCN_DHID * 2);     // [8192][256]
    bf16_t* X2T    = (bf16_t*)alloc((size_t)GCN_DOUT * GCN_N * 2);     // [64][8192]
    float*  logits = (float*)alloc((size_t)GCN_N * GCN_DOUT * 4);      // [8192][64]

    const dim3 blk(256);

    // Weight transposes -> bf16 (B operands consumed column-major).
    transpose_to_bf16<<<(GCN_DIN * GCN_DHID + 255) / 256, blk, 0, stream>>>(
        W1, W1T, GCN_DIN, GCN_DHID);
    transpose_to_bf16<<<(GCN_DHID * GCN_DOUT + 255) / 256, blk, 0, stream>>>(
        W2, W2T, GCN_DHID, GCN_DOUT);

    // K1: X1T = (feats @ W1)^T     [block 64x256, feats read once, 48KB LDS]
    wmma_gemm<8, 2, 32, true, 0><<<dim3(1, GCN_N / 64), blk, 49152, stream>>>(
        feats, W1T, nullptr, X1T, GCN_DIN, GCN_N);

    // K2: H = relu(adj @ X1 + b1)  [block 64x256, adj read once, 48KB LDS]
    wmma_gemm<8, 2, 32, true, 1><<<dim3(1, GCN_N / 64), blk, 49152, stream>>>(
        adj, X1T, b1, H, GCN_N, GCN_DHID);

    // K3: X2T = (H @ W2)^T         [block 64x64, 24KB LDS]
    wmma_gemm<2, 2, 64, false, 0><<<dim3(1, GCN_N / 64), blk, 24576, stream>>>(
        H, W2T, nullptr, X2T, GCN_DHID, GCN_N);

    // K4: logits = adj @ X2 + b2   [block 64x64, adj read once, 24KB LDS]
    wmma_gemm<2, 2, 64, true, 2><<<dim3(1, GCN_N / 64), blk, 24576, stream>>>(
        adj, X2T, b2, logits, GCN_N, GCN_DOUT);

    // K5: log_softmax rows
    log_softmax_rows<<<GCN_N, 32, 0, stream>>>(logits, (float*)d_out);
}
